// GeneralizedSupervisedNTXenLoss_23742579213066
// MI455X (gfx1250) — matrix operand order repaired and checked by reading, not compiled, not run on hardware
//
#include <hip/hip_runtime.h>

// GeneralizedSupervisedNTXenLoss, fused for MI455X (gfx1250, wave32, WMMA).
//
// Loss reduced to a single Gram matrix G = Zn @ Zn^T over Zn = l2norm([z_i; z_j])
// (8192 x 256), consumed tile-by-tile (never materialized). Per row r (diag
// INCLUDED during accumulation, subtracted analytically in finalize):
//   s2 = sum_c exp(2*G[r,c] - 2)      (sup branch, TEMP_SUP=0.5; |G|<=1 -> fixed shift)
//   s1 = sum_c e2^5 = sum_c exp(10*G[r,c] - 10)   (pure branch, TEMP=0.1)
//   nrm = sum_c w_rc, ws = sum_c w_rc * 2*G[r,c], w_rc = exp(-0.5*(l_r-l_c)^2)
// finalize: subtract diag (a_rr = f16 self-dot, w_rr = 1), then
// loss = (0.5/N) * sum_r [ (10+log s1) - 10*G[r,partner] ]
//      - (0.5/N) * sum_r [ ws/nrm - (2+log s2) + log nrm ]
//
// ws layout: [ Zn f16 (2N*256) | S1 | S2 | NRM | WS ]  (~4.3 MB)

typedef __attribute__((ext_vector_type(16))) _Float16 v16h;
typedef __attribute__((ext_vector_type(8)))  float    v8f;
struct alignas(16) U4 { unsigned int x, y, z, w; };

#define D 256          // feature dim (fixed by reference)
#define MAXN 4096      // N (labels cached once, indexed mod N)
#define BPITCH 264     // LDS B row pitch in halves (256 + 8 pad -> no bank conflicts)

// ---------------- zero accumulators + output ----------------
__global__ void ntx_zero_kernel(float* __restrict__ acc, int n, float* __restrict__ out) {
  int i = blockIdx.x * blockDim.x + threadIdx.x;
  if (i < n) acc[i] = 0.0f;
  if (i == 0) *out = 0.0f;
}

// ---------------- normalize rows -> f16 Zn ----------------
__global__ __launch_bounds__(256)
void ntx_prep_kernel(const float* __restrict__ zi, const float* __restrict__ zj,
                     _Float16* __restrict__ Zn, int N) {
  const int r = blockIdx.x;          // row of [z_i ; z_j]
  const int t = threadIdx.x;         // 256 threads == D
  const float* src = (r < N) ? (zi + (size_t)r * D) : (zj + (size_t)(r - N) * D);
  float x = src[t];
  float ss = x * x;
  #pragma unroll
  for (int off = 16; off >= 1; off >>= 1) ss += __shfl_xor(ss, off, 32);
  __shared__ float red[8];
  if ((t & 31) == 0) red[t >> 5] = ss;
  __syncthreads();
  float tot = red[0] + red[1] + red[2] + red[3] + red[4] + red[5] + red[6] + red[7];
  float inv = 1.0f / fmaxf(sqrtf(tot), 1e-12f);
  Zn[(size_t)r * D + t] = (_Float16)(x * inv);
}

// ---------------- fused Gram + online reductions ----------------
// block = 256 threads = 8 waves; wave w owns rows [blockIdx.x*128 + w*16, +16)
// blockIdx.y selects a column chunk (grid.y = CSPLIT); partials -> atomicAdd.
// 32-column super-tiles, double-buffered in LDS (stage t+1 while computing t,
// 1 barrier/iter). TWO independent WMMA chains (c0, c1) per iteration so the
// scheduler can overlap one chain's ds_loads with the other chain's WMMAs.
__global__ __launch_bounds__(256)
void ntx_main_kernel(const _Float16* __restrict__ Zn, const float* __restrict__ labels,
                     float* __restrict__ S1, float* __restrict__ S2,
                     float* __restrict__ NRM, float* __restrict__ WS,
                     int N, int twoN, int colsPerChunk) {
  __shared__ __align__(16) _Float16 ldsB[2][32 * BPITCH]; // 2 x (32 cols x 256 K + pad)
  __shared__ float ldsLab[MAXN];                          // labels (index mod N)

  const int t    = threadIdx.x;
  const int lane = t & 31;
  const int wave = t >> 5;
  const int m    = lane & 15;   // M (A) / N (B,C) index within tile
  const int hi   = lane >> 4;   // half-wave select
  const int rowA = blockIdx.x * 128 + wave * 16;
  const int col0 = blockIdx.y * colsPerChunk;
  const int nPair = colsPerChunk >> 5;   // 32-col super-tiles

  // labels into LDS (single copy; duplication handled by mod-N indexing)
  for (int i = t; i < N; i += 256) ldsLab[i] = labels[i];

  // A fragments: 16 rows x 256 K (8 chunks of K=32), per ISA 16-bit A layout:
  // lanes 0-15: M=lane, halves K={kc*32+0..7, kc*32+16..23};
  // lanes16-31: M=lane-16, K={kc*32+8..15, kc*32+24..31}.
  v16h afrag[8];
  {
    const _Float16* arow = Zn + (size_t)(rowA + m) * D;
    #pragma unroll
    for (int kc = 0; kc < 8; ++kc) {
      int k0 = kc * 32 + hi * 8;
      ((U4*)&afrag[kc])[0] = *(const U4*)(arow + k0);
      ((U4*)&afrag[kc])[1] = *(const U4*)(arow + k0 + 16);
    }
  }

  // cooperative super-tile stage: 32 cols x 256 halves = 16 KB, 64B/thread
  const int br = t >> 3;               // staged row 0..31
  const int bs = (t & 7) * 32;         // 32-half (64B) segment within row
  const U4* gsrc0 = (const U4*)(Zn + (size_t)(col0 + br) * D + bs);

  {  // prologue: stage super-tile 0 into buffer 0
    U4* dst = (U4*)(ldsB[0] + br * BPITCH + bs);
    #pragma unroll
    for (int q = 0; q < 4; ++q) dst[q] = gsrc0[q];
  }
  __syncthreads();

  float labR[8];
  #pragma unroll
  for (int v = 0; v < 8; ++v) {
    int rr = rowA + v + hi * 8;
    labR[v] = ldsLab[rr - (rr >= N ? N : 0)];
  }

  float s1a[8] = {0,0,0,0,0,0,0,0}, s2a[8] = {0,0,0,0,0,0,0,0};
  float nra[8] = {0,0,0,0,0,0,0,0}, wsa[8] = {0,0,0,0,0,0,0,0};

  for (int it = 0; it < nPair; ++it) {
    const int cur = it & 1;
    // stage next super-tile into the other buffer (overlaps with compute below)
    if (it + 1 < nPair) {
      const U4* src = gsrc0 + (size_t)(it + 1) * 32 * (D / 8); // +32 rows of 256 halves
      U4* dst = (U4*)(ldsB[1 - cur] + br * BPITCH + bs);
      #pragma unroll
      for (int q = 0; q < 4; ++q) dst[q] = src[q];
      if (it + 2 < nPair)
        __builtin_prefetch((const void*)(gsrc0 + (size_t)(it + 2) * 32 * (D / 8)), 0, 0);
    }

    // two independent 16x16 Gram tiles (cols [0,16) and [16,32) of super-tile)
    const _Float16* brow0 = ldsB[cur] + m * BPITCH + hi * 8;
    const _Float16* brow1 = brow0 + 16 * BPITCH;
    v8f c0 = {}, c1 = {};
    #pragma unroll
    for (int kc = 0; kc < 8; ++kc) {
      v16h b0, b1;
      ((U4*)&b0)[0] = *(const U4*)(brow0 + kc * 32);
      ((U4*)&b0)[1] = *(const U4*)(brow0 + kc * 32 + 16);
      ((U4*)&b1)[0] = *(const U4*)(brow1 + kc * 32);
      ((U4*)&b1)[1] = *(const U4*)(brow1 + kc * 32 + 16);
      c0 = __builtin_amdgcn_wmma_f32_16x16x32_f16(
               false, afrag[kc], false, b0, (short)0, c0, false, false);
      c1 = __builtin_amdgcn_wmma_f32_16x16x32_f16(
               false, afrag[kc], false, b1, (short)0, c1, false, false);
    }

    // consume both tiles unconditionally (diag folded in; removed in finalize):
    // C layout -> VGPR v, lanes[0:16)=M=v, lanes[16:32)=M=v+8, N=lane%16
    const int gc0 = col0 + it * 32 + m;
    const int gc1 = gc0 + 16;
    const float labC0 = ldsLab[gc0 - (gc0 >= N ? N : 0)];
    const float labC1 = ldsLab[gc1 - (gc1 >= N ? N : 0)];
    #pragma unroll
    for (int v = 0; v < 8; ++v) {
      {
        const float a  = c0[v];
        const float e2 = __expf(2.0f * a - 2.0f);   // 1 TRANS
        const float e2sq = e2 * e2;
        const float e1 = e2sq * e2sq * e2;          // exp(10a-10) = e2^5, VALU only
        const float dl = labR[v] - labC0;
        const float w  = __expf(-0.5f * dl * dl);   // 1 TRANS
        s1a[v] += e1; s2a[v] += e2; nra[v] += w; wsa[v] += w * (2.0f * a);
      }
      {
        const float a  = c1[v];
        const float e2 = __expf(2.0f * a - 2.0f);
        const float e2sq = e2 * e2;
        const float e1 = e2sq * e2sq * e2;
        const float dl = labR[v] - labC1;
        const float w  = __expf(-0.5f * dl * dl);
        s1a[v] += e1; s2a[v] += e2; nra[v] += w; wsa[v] += w * (2.0f * a);
      }
    }
    __syncthreads(); // next iter reads buffer staged above / re-stages this one
  }

  // reduce each row across its 16 lanes (xor stays inside each half-wave)
  #pragma unroll
  for (int v = 0; v < 8; ++v) {
    #pragma unroll
    for (int off = 1; off < 16; off <<= 1) {
      s1a[v] += __shfl_xor(s1a[v], off, 32);
      s2a[v] += __shfl_xor(s2a[v], off, 32);
      nra[v] += __shfl_xor(nra[v], off, 32);
      wsa[v] += __shfl_xor(wsa[v], off, 32);
    }
  }
  if (m == 0) { // lanes 0 (rows v) and 16 (rows v+8)
    #pragma unroll
    for (int v = 0; v < 8; ++v) {
      const int r = rowA + v + hi * 8;
      atomicAdd(&S1[r],  s1a[v]);
      atomicAdd(&S2[r],  s2a[v]);
      atomicAdd(&NRM[r], nra[v]);
      atomicAdd(&WS[r],  wsa[v]);
    }
  }
}

// ---------------- per-row logs + partner dot + diag removal + scalar reduce ----------------
__global__ __launch_bounds__(256)
void ntx_final_kernel(const _Float16* __restrict__ Zn,
                      const float* __restrict__ S1, const float* __restrict__ S2,
                      const float* __restrict__ NRM, const float* __restrict__ WS,
                      float* __restrict__ out, int N, int twoN) {
  const int r = blockIdx.x * blockDim.x + threadIdx.x;
  if (r >= twoN) return;
  const int p = (r < N) ? (r + N) : (r - N);
  const _Float16* a = Zn + (size_t)r * D;
  const _Float16* b = Zn + (size_t)p * D;
  float pos = 0.0f, sd = 0.0f;       // partner dot, self dot (diag value in f16 arith)
  #pragma unroll 8
  for (int k = 0; k < D; ++k) {
    const float av = (float)a[k];
    pos += av * (float)b[k];
    sd  += av * av;
  }

  // remove diagonal contributions (same e2^5 identity as the main kernel)
  const float e2d  = __expf(2.0f * sd - 2.0f);
  const float e2ds = e2d * e2d;
  const float e1d  = e2ds * e2ds * e2d;
  const float s1  = S1[r]  - e1d;
  const float s2  = S2[r]  - e2d;
  const float nrm = NRM[r] - 1.0f;          // w_rr = exp(0) = 1 exactly
  const float ws  = WS[r]  - 2.0f * sd;

  const float lse1 = 10.0f + __logf(s1);
  const float lse2 =  2.0f + __logf(s2);
  const float invN = 1.0f / (float)N;
  const float contrib = 0.5f * invN * (lse1 - 10.0f * pos)
                      - 0.5f * invN * (ws / nrm - lse2 + __logf(nrm));
  atomicAdd(out, contrib);
}

extern "C" void kernel_launch(void* const* d_in, const int* in_sizes, int n_in,
                              void* d_out, int out_size, void* d_ws, size_t ws_size,
                              hipStream_t stream) {
  const float* zi     = (const float*)d_in[0];
  const float* zj     = (const float*)d_in[1];
  const float* labels = (const float*)d_in[2];
  const int N    = in_sizes[2];   // 4096
  const int twoN = 2 * N;         // 8192

  _Float16* Zn = (_Float16*)d_ws;
  float* S1  = (float*)((char*)d_ws + (size_t)twoN * D * sizeof(_Float16));
  float* S2  = S1  + twoN;
  float* NRM = S2  + twoN;
  float* WS  = NRM + twoN;
  float* out = (float*)d_out;

  ntx_zero_kernel<<<(4 * twoN + 255) / 256, 256, 0, stream>>>(S1, 4 * twoN, out);
  ntx_prep_kernel<<<twoN, 256, 0, stream>>>(zi, zj, Zn, N);

  const int CSPLIT = 8;                       // 64 row-tiles x 8 col-chunks = 512 WGs
  const int colsPerChunk = twoN / CSPLIT;     // 1024 cols -> 32 super-tiles
  dim3 grid(twoN / 128, CSPLIT);
  ntx_main_kernel<<<grid, 256, 0, stream>>>(Zn, labels, S1, S2, NRM, WS, N, twoN, colsPerChunk);

  ntx_final_kernel<<<(twoN + 255) / 256, 256, 0, stream>>>(Zn, S1, S2, NRM, WS, out, N, twoN);
}
